// DistanceTransformLayer_66305705116155
// MI455X (gfx1250) — compile-verified, matrix operand order checked by reading, not run on hardware
//
#include <hip/hip_runtime.h>
#include <hip/hip_bf16.h>

typedef __attribute__((ext_vector_type(16))) _Float16 v16h;
typedef __attribute__((ext_vector_type(8)))  float    v8f;

#define DT_H   128
#define DT_W   128
#define NPIX   (DT_H * DT_W)        // 16384
#define NTILES (NPIX / 16)          // 1024 tiles of 16 pixels

// ---------------------------------------------------------------------------
// Kernel 1: pool boundary mask over batch; emit candidate coords + norms.
//   qx[q] = h, qy[q] = w, qn[q] = h*h + w*w  (boundary)  or  +inf (not)
// ---------------------------------------------------------------------------
__global__ void dt_prep(const float* __restrict__ f, int nb,
                        float* __restrict__ qx, float* __restrict__ qy,
                        float* __restrict__ qn)
{
    int q = blockIdx.x * blockDim.x + threadIdx.x;
    if (q >= NPIX) return;
    bool bnd = false;
    for (int b = 0; b < nb; ++b)
        bnd |= (f[b * NPIX + q] > 0.5f);
    int h = q >> 7;
    int w = q & (DT_W - 1);
    qx[q] = (float)h;
    qy[q] = (float)w;
    qn[q] = bnd ? (float)(h * h + w * w) : __builtin_inff();
}

// ---------------------------------------------------------------------------
// Kernel 2: pairwise squared distances via WMMA Gram trick + running min.
// One wave (32 lanes) per 16-output-pixel tile; loops over 1024 candidate
// tiles, one v_wmma_f32_16x16x32_f16 per candidate tile.
//
//   C[M][N] = i_M*h_N + j_M*w_N     (exact: f16 ints <=127, f32 accumulate)
//   d2      = (i_M^2 + j_M^2) + qn[N] - 2*C[M][N]
// ---------------------------------------------------------------------------
__global__ void __launch_bounds__(256)
dt_wmma(const float* __restrict__ qx, const float* __restrict__ qy,
        const float* __restrict__ qn, float* __restrict__ dist)
{
    const int lane = threadIdx.x & 31;
    const int wave = threadIdx.x >> 5;
    const int tile = blockIdx.x * (blockDim.x >> 5) + wave;   // 0..NTILES-1
    const int hi   = lane >> 4;                                // half-wave id
    const int col  = lane & 15;                                // M (A) / N (B)

    // Output pixels covered by this tile: p = 16*tile + M, M = 0..15.
    // 16 | 128 => row index i is constant within a tile; j = j0 + M.
    const int p0   = tile * 16;
    const int iRow = p0 >> 7;
    const int j0   = p0 & (DT_W - 1);

    __builtin_prefetch(qn, 0, 0);  // global_prefetch_b8

    // A fragment (16x32 f16): lanes 0-15 hold K=0..7 in elems 0..7.
    // Only K=0 (=i) and K=1 (=j) are non-zero; lanes 16-31 hold K=8..15 -> 0.
    v16h a;
    #pragma unroll
    for (int k = 0; k < 16; ++k) a[k] = (_Float16)0;
    if (hi == 0) {
        a[0] = (_Float16)(float)iRow;
        a[1] = (_Float16)(float)(j0 + col);
    }

    // Per-lane output-row norms: acc[r] is row M = r + 8*hi, col N = col.
    float pn[8];
    #pragma unroll
    for (int r = 0; r < 8; ++r) {
        int jr = j0 + r + hi * 8;
        pn[r] = (float)(iRow * iRow + jr * jr);
    }

    float acc[8];
    #pragma unroll
    for (int r = 0; r < 8; ++r) acc[r] = __builtin_inff();

    for (int ct = 0; ct < NTILES; ++ct) {
        const int q = ct * 16 + col;

        // B fragment (32x16 f16): lane = column N; K=0 (=h), K=1 (=w) in
        // elems 0,1 of lanes 0-15; lanes 16-31 (K=8..15,24..31) are zero.
        float bx = qx[q];
        float by = qy[q];
        if (hi) { bx = 0.0f; by = 0.0f; }     // v_cndmask, no divergence
        v16h b;
        #pragma unroll
        for (int k = 0; k < 16; ++k) b[k] = (_Float16)0;
        b[0] = (_Float16)bx;
        b[1] = (_Float16)by;

        const float qncol = qn[q];            // +inf kills non-boundary cols

        v8f c = {};
        c = __builtin_amdgcn_wmma_f32_16x16x32_f16(
                /*neg_a=*/false, a, /*neg_b=*/false, b,
                /*c_mod=*/(short)0, c, /*reuse_a=*/false, /*reuse_b=*/false);

        #pragma unroll
        for (int r = 0; r < 8; ++r) {
            float d2 = (pn[r] + qncol) - 2.0f * c[r];
            acc[r] = fminf(acc[r], d2);
        }
    }

    // Min across the 16 lanes of each half-wave (same M, different N%16).
    #pragma unroll
    for (int r = 0; r < 8; ++r) {
        float m = acc[r];
        m = fminf(m, __shfl_xor(m, 1, 32));
        m = fminf(m, __shfl_xor(m, 2, 32));
        m = fminf(m, __shfl_xor(m, 4, 32));
        m = fminf(m, __shfl_xor(m, 8, 32));
        acc[r] = m;
    }

    if (col == 0) {
        #pragma unroll
        for (int r = 0; r < 8; ++r) {
            int M = r + hi * 8;
            dist[p0 + M] = sqrtf(fmaxf(acc[r], 0.0f));
        }
    }
}

// ---------------------------------------------------------------------------
// Kernel 3: broadcast per-pixel distance to all batch elements.
// ---------------------------------------------------------------------------
__global__ void dt_bcast(const float* __restrict__ dist,
                         float* __restrict__ out, int n)
{
    int t = blockIdx.x * blockDim.x + threadIdx.x;
    if (t < n) out[t] = dist[t & (NPIX - 1)];
}

// ---------------------------------------------------------------------------
extern "C" void kernel_launch(void* const* d_in, const int* in_sizes, int n_in,
                              void* d_out, int out_size, void* d_ws, size_t ws_size,
                              hipStream_t stream)
{
    const float* f  = (const float*)d_in[0];
    float* out      = (float*)d_out;
    const int nb    = in_sizes[0] / NPIX;   // batch*channel copies (8)

    // Workspace layout (floats): qx | qy | qn | dist  = 256 KB total.
    float* qx   = (float*)d_ws;
    float* qy   = qx + NPIX;
    float* qn   = qy + NPIX;
    float* dist = qn + NPIX;

    dt_prep<<<(NPIX + 255) / 256, 256, 0, stream>>>(f, nb, qx, qy, qn);

    // 1024 tiles, 8 waves per 256-thread block -> 128 blocks.
    dt_wmma<<<NTILES / 8, 256, 0, stream>>>(qx, qy, qn, dist);

    dt_bcast<<<(out_size + 255) / 256, 256, 0, stream>>>(dist, out, out_size);
}
